// CausalHierarchicalMemoryLM_52063593562188
// MI455X (gfx1250) — matrix operand order repaired and checked by reading, not compiled, not run on hardware
//
#include <hip/hip_runtime.h>
#include <math.h>

// Problem constants (from reference setup_inputs)
#define BB 16
#define SS 2048
#define DD 512
#define MM 256
#define RR 64
#define TK 16
#define LRS 0.1f
#define SMASS 4.0f
#define LNEPS 1e-5f

typedef __attribute__((ext_vector_type(16))) __bf16 v16bf;
typedef __attribute__((ext_vector_type(8)))  float  v8f;
typedef __attribute__((ext_vector_type(4)))  unsigned int u32x4;
typedef __attribute__((ext_vector_type(4)))  int    i32x4;
typedef __attribute__((ext_vector_type(8)))  int    i32x8;

// ---- ISA-derived lane layouts (CDNA5 ISA §7.12.2) -------------
// bf16 A 16x32: lane L: M=L&15, half=L>>4. element e (pair v=e>>1):
//   k = (v<4 ? 2v : 16+2(v-4)) + (e&1) + 8*half
__device__ __forceinline__ int a_pair_base(int v) {
    return (v < 4) ? (2 * v) : (16 + 2 * (v - 4));
}
// bf16 B 32x16: lane L: N=L&15, half=L>>4, ISA k order as A; code uses k = e + 16*half for A and B
// f32 C/D 16x16: lane L: N=L&15, element v: M = v + 8*(L>>4)

__device__ __forceinline__ void pack4(v16bf& v, int base, float4 q) {
    v[base + 0] = (__bf16)q.x;
    v[base + 1] = (__bf16)q.y;
    v[base + 2] = (__bf16)q.z;
    v[base + 3] = (__bf16)q.w;
}

__device__ __forceinline__ float block_reduce_sum(float v, float* red, int tid) {
    red[tid] = v;
    __syncthreads();
    for (int off = 128; off > 0; off >>= 1) {
        if (tid < off) red[tid] += red[tid + off];
        __syncthreads();
    }
    float r = red[0];
    __syncthreads();
    return r;
}

// ---------------------------------------------------------------------------
// TDM: async 2D tensor tile (f32) Global -> LDS (CDNA5 ISA)
// D# group0: [1:0]=count=1, [63:32]=lds_addr, [120:64]=global_addr, [127:126]=type=2
// D# group1: [17:16]=data_size(2 -> 4B), [20]=pad_enable, [24:22]=pad_interval,
//            [31:25]=pad_amount, [79:48]=tensor_dim0, [111:80]=tensor_dim1,
//            [127:112]=tile_dim0, [143:128]=tile_dim1, [207:160]=dim0_stride.
// pad_interval=5 (every 64 DWORDs) + pad_amount=0 (1 DWORD) => LDS row stride
// of 65 DWORDs, which makes the two half-wave K-groups bank-conflict free.
// ---------------------------------------------------------------------------
#define WP_STRIDE 65
#define WP_BYTES  (64 * WP_STRIDE * 4)   /* one 64x64 f32 panel incl. padding */

__device__ __forceinline__ void tdm_load_w_panel(unsigned lds_byte_off,
                                                 const float* gsrc,
                                                 unsigned tensor_rows) {
    unsigned long long ga = (unsigned long long)(const void*)gsrc;
    u32x4 g0 = {0u, 0u, 0u, 0u};
    g0[0] = 1u;                                    // count=1 (user mode)
    g0[1] = lds_byte_off;                          // lds_addr
    g0[2] = (unsigned)(ga & 0xFFFFFFFFull);
    g0[3] = (unsigned)((ga >> 32) & 0x1FFFFFFull) | (2u << 30);  // type=2
    i32x8 g1 = {0, 0, 0, 0, 0, 0, 0, 0};
    g1[0] = (int)((2u << 16) | (1u << 20) | (5u << 22));  // 4B, pad 1 DW / 64 DW
    g1[1] = (int)(64u << 16);                      // tensor_dim0 = 64 (lo16)
    g1[2] = (int)((tensor_rows & 0xFFFFu) << 16);  // dim0 hi16=0 | tensor_dim1 lo16
    g1[3] = (int)(((tensor_rows >> 16) & 0xFFFFu) | (64u << 16));  // dim1 hi | tile_dim0
    g1[4] = (int)64;                               // tile_dim1 = 64 rows
    g1[5] = (int)64;                               // tensor_dim0_stride = 64 elems
    i32x4 z4 = {0, 0, 0, 0};
#if __clang_major__ >= 23
    i32x8 z8 = {0, 0, 0, 0, 0, 0, 0, 0};
    __builtin_amdgcn_tensor_load_to_lds(g0, g1, z4, z4, z8, 0);
#else
    __builtin_amdgcn_tensor_load_to_lds(g0, g1, z4, z4, 0);
#endif
}

// ---------------------------------------------------------------------------
// LN of rows of a [nRows, 512] matrix (used for mem_val0 = LN(init_val))
// ---------------------------------------------------------------------------
__global__ __launch_bounds__(256) void ln_rows(const float* __restrict__ X,
                                               const float* __restrict__ g,
                                               const float* __restrict__ bvec,
                                               float* __restrict__ Y) {
    const int m = blockIdx.x, tid = threadIdx.x;
    __shared__ float red[256];
    __shared__ float sMu, sRs;
    float a0 = X[(size_t)m * DD + tid];
    float a1 = X[(size_t)m * DD + 256 + tid];
    float sum = block_reduce_sum(a0 + a1, red, tid);
    if (tid == 0) sMu = sum * (1.0f / DD);
    __syncthreads();
    const float mu = sMu;
    const float d0 = a0 - mu, d1 = a1 - mu;
    float s2 = block_reduce_sum(d0 * d0 + d1 * d1, red, tid);
    if (tid == 0) sRs = rsqrtf(s2 * (1.0f / DD) + LNEPS);
    __syncthreads();
    const float rs = sRs;
    Y[(size_t)m * DD + tid]       = d0 * rs * g[tid]       + bvec[tid];
    Y[(size_t)m * DD + 256 + tid] = d1 * rs * g[256 + tid] + bvec[256 + tid];
}

// ---------------------------------------------------------------------------
// signed-abs softmax * STATE_MASS over a 256-row (one block per batch row)
// ---------------------------------------------------------------------------
__global__ __launch_bounds__(256) void signed_softmax_mass(const float* __restrict__ raw,
                                                           float* __restrict__ out) {
    const int b = blockIdx.x, tid = threadIdx.x;
    __shared__ float red[256];
    float x = raw[(size_t)b * MM + tid];
    if (!(x == x)) x = 0.0f;  // nan_to_num
    const float a = fabsf(x);
    red[tid] = a;
    __syncthreads();
    for (int off = 128; off > 0; off >>= 1) {
        if (tid < off) red[tid] = fmaxf(red[tid], red[tid + off]);
        __syncthreads();
    }
    const float mx = red[0];
    __syncthreads();
    const float e = __expf(a - mx);
    const float sum = block_reduce_sum(e, red, tid);
    const float sgn = (x > 0.0f) ? 1.0f : ((x < 0.0f) ? -1.0f : 0.0f);
    out[(size_t)b * MM + tid] = sgn * e / sum * SMASS;
}

// ---------------------------------------------------------------------------
// One K-panel (64 deep) of the projection MMA: A from global X row, B from the
// TDM-staged LDS panel (stride 65 DWORDs -> conflict-free ds_load_b32).
// NOTE: loads are intentionally non-volatile so InferAddressSpaces lowers
// them to ds_load_b32 (volatile forced flat_load in round 3). Poison-folding
// of "never-stored LDS" is defeated in the caller by escaping the buffer
// address into an inline asm with a memory clobber after s_wait_tensorcnt.
// ---------------------------------------------------------------------------
__device__ __forceinline__ void panel_mma(const float* __restrict__ xrow,
                                          const float* wp, int half, int ln,
                                          v8f acc[4]) {
#pragma unroll
    for (int kk = 0; kk < 64; kk += 32) {
        v16bf A;
#pragma unroll
        for (int v = 0; v < 8; ++v) {
            const float2 p = *(const float2*)(xrow + kk + 2 * v + 16 * half);
            A[2 * v]     = (__bf16)p.x;
            A[2 * v + 1] = (__bf16)p.y;
        }
#pragma unroll
        for (int t = 0; t < 4; ++t) {
            v16bf Bv;
#pragma unroll
            for (int e = 0; e < 16; ++e) {
                const int k = kk + e + 16 * half;
                Bv[e] = (__bf16)wp[k * WP_STRIDE + t * 16 + ln];
            }
            acc[t] = __builtin_amdgcn_wmma_f32_16x16x32_bf16(
                false, A, false, Bv, (short)0, acc[t], false, false);
        }
    }
}

// ---------------------------------------------------------------------------
// Y[row, r] = (X[row,:512] @ W[:512, r] + bias[r]) * extraScale
//             * (colScale? colScale[r] : 1) * (rowScale? rowScale[row] : 1)
// One wave per block computes a 16-row x 64-col tile. W is streamed into LDS
// by the Tensor Data Mover in double-buffered 64-deep K-panels (TENSORcnt
// pipelining: issue p+1, wait<=1, consume p, drain DS before reuse).
// ---------------------------------------------------------------------------
__global__ __launch_bounds__(32) void proj_gemm(const float* __restrict__ X,
                                                const float* __restrict__ W,
                                                const float* __restrict__ bias,
                                                const float* __restrict__ colScale,
                                                const float* __restrict__ rowScale,
                                                float extraScale,
                                                float* __restrict__ Y) {
    // Only shared allocation in this kernel -> static LDS base offset 0.
    __shared__ float wlds[2][64 * WP_STRIDE];

    const int row0 = blockIdx.x * 16;
    const int lane = threadIdx.x;
    const int half = lane >> 4, ln = lane & 15;

    v8f acc[4] = {};
    const float* xrow = X + (size_t)(row0 + ln) * DD;

    tdm_load_w_panel(0u, W, DD);                          // panel 0 -> buf 0
    for (int p = 0; p < 7; ++p) {
        tdm_load_w_panel((unsigned)(((p + 1) & 1) * WP_BYTES),
                         W + (size_t)(p + 1) * 64 * RR, DD);  // panel p+1
        __builtin_amdgcn_s_wait_tensorcnt(1);             // panel p resident
        const float* wp = &wlds[p & 1][0];
        // Escape the LDS address into asm + memory clobber: models the TDM
        // write so LLVM keeps the ds_loads (never-stored LDS would fold to
        // poison), while the clean GEP chain still infers addrspace(3).
        asm volatile("" ::"v"(wp) : "memory");
        panel_mma(xrow + (size_t)p * 64, wp, half, ln, acc);
        // drain our LDS reads before the TDM may overwrite this buffer
        asm volatile("s_wait_dscnt 0x0" ::: "memory");
    }
    __builtin_amdgcn_s_wait_tensorcnt(0);                 // last panel resident
    {
        const float* wp = &wlds[1][0];
        asm volatile("" ::"v"(wp) : "memory");
        panel_mma(xrow + (size_t)7 * 64, wp, half, ln, acc);
    }

#pragma unroll
    for (int t = 0; t < 4; ++t) {
        const int col = t * 16 + ln;
        const float cs = extraScale * (colScale ? colScale[col] : 1.0f);
        const float bz = bias ? bias[col] : 0.0f;
#pragma unroll
        for (int v = 0; v < 8; ++v) {
            const int row = row0 + v + 8 * half;
            const float rsc = rowScale ? rowScale[row] : 1.0f;
            Y[(size_t)row * RR + col] = (acc[t][v] + bz) * cs * rsc;
        }
    }
}

// ---------------------------------------------------------------------------
// Fused: score tile (16 dst rows x 16 src cols) via 2x WMMA bf16 (K=R=64),
// running per-row top-16 by |score| in registers, then edge signed softmax.
// grid = (M/16, B); one wave per block. Never materializes [B,M,S] scores.
// ---------------------------------------------------------------------------
__global__ __launch_bounds__(32) void score_topk(const float* __restrict__ ptMat, long ptBS,
                                                 const float* __restrict__ psMat, long psBS,
                                                 int nSrc,
                                                 int* __restrict__ outIdx,
                                                 float* __restrict__ outEdg) {
    const int mt = blockIdx.x, b = blockIdx.y;
    const int lane = threadIdx.x;
    const int half = lane >> 4, ln = lane & 15;
    const int m0 = mt * 16;

    // A = pt rows [16 x 64] (persistent in registers)
    const float* prow = ptMat + (size_t)b * ptBS + (size_t)(m0 + ln) * RR;
    v16bf A0, A1;
#pragma unroll
    for (int e = 0; e < 16; ++e) {
        const int kk = e + 16 * half;
        A0[e] = (__bf16)prow[kk];
        A1[e] = (__bf16)prow[32 + kk];
    }

    __shared__ float sc[16 * 17];

    float sval[TK], sabs[TK];
    int   sidx[TK];
#pragma unroll
    for (int j = 0; j < TK; ++j) { sval[j] = 0.0f; sabs[j] = -1.0f; sidx[j] = 0; }
    float minAbs = -1.0f;
    int   minPos = 0;

    for (int s0 = 0; s0 < nSrc; s0 += 16) {
        // B tile: B[k=r][n=s] = ps[s, r] (transposed read, contiguous in r)
        const float* srow = psMat + (size_t)b * psBS + (size_t)(s0 + ln) * RR + 16 * half;
        const float4* q0 = (const float4*)srow;
        const float4* q1 = (const float4*)(srow + 32);
        v16bf B0, B1;
        pack4(B0, 0, q0[0]); pack4(B0, 4, q0[1]); pack4(B0, 8, q0[2]); pack4(B0, 12, q0[3]);
        pack4(B1, 0, q1[0]); pack4(B1, 4, q1[1]); pack4(B1, 8, q1[2]); pack4(B1, 12, q1[3]);

        v8f acc = {};
        acc = __builtin_amdgcn_wmma_f32_16x16x32_bf16(false, A0, false, B0, (short)0, acc, false, false);
        acc = __builtin_amdgcn_wmma_f32_16x16x32_bf16(false, A1, false, B1, (short)0, acc, false, false);

#pragma unroll
        for (int v = 0; v < 8; ++v) sc[(v + 8 * half) * 17 + ln] = acc[v];
        __syncthreads();

        if (half == 0) {  // lanes 0..15 each own dst row m0+ln
#pragma unroll
            for (int n = 0; n < 16; ++n) {
                const float val = sc[ln * 17 + n];
                const float a = fabsf(val);
                if (a > minAbs) {
#pragma unroll
                    for (int j = 0; j < TK; ++j)
                        if (j == minPos) { sval[j] = val; sabs[j] = a; sidx[j] = s0 + n; }
                    minAbs = sabs[0]; minPos = 0;
#pragma unroll
                    for (int j = 1; j < TK; ++j)
                        if (sabs[j] < minAbs) { minAbs = sabs[j]; minPos = j; }
                }
            }
        }
        __syncthreads();
    }

    if (half == 0) {
        float mx = sabs[0];
#pragma unroll
        for (int j = 1; j < TK; ++j) mx = fmaxf(mx, sabs[j]);
        float ev[TK], sum = 0.0f;
#pragma unroll
        for (int j = 0; j < TK; ++j) { ev[j] = __expf(sabs[j] - mx); sum += ev[j]; }
        const float inv = 1.0f / sum;
        const size_t o = (((size_t)b * MM) + m0 + ln) * TK;
#pragma unroll
        for (int j = 0; j < TK; ++j) {
            const float sg = (sval[j] > 0.0f) ? 1.0f : ((sval[j] < 0.0f) ? -1.0f : 0.0f);
            outEdg[o + j] = sg * ev[j] * inv;
            outIdx[o + j] = sidx[j];
        }
    }
}

// ---------------------------------------------------------------------------
// out[b,m,:] = LN(base[b,m,:] + sum_k edg_k * src[b, idx_k, :]) ; optional
// state raw update. One 256-thread block per (b,m); 2 channels per thread.
// ---------------------------------------------------------------------------
__global__ __launch_bounds__(256) void apply_ln(const float* __restrict__ baseVal, long baseBS,
                                                const float* __restrict__ srcVal, long srcBS,
                                                const float* __restrict__ baseState, long baseStBS,
                                                const float* __restrict__ srcState, long srcStBS,
                                                const int* __restrict__ idx,
                                                const float* __restrict__ edg,
                                                const float* __restrict__ g,
                                                const float* __restrict__ bvec,
                                                float* __restrict__ outVal,
                                                float* __restrict__ outState) {
    const int bm = blockIdx.x;
    const int b = bm >> 8, m = bm & 255;
    const int tid = threadIdx.x;
    __shared__ int   sIdx[TK];
    __shared__ float sW[TK], sSt[TK];
    __shared__ float red[256];
    __shared__ float sMu, sRs;

    if (tid < TK) {
        const int k = idx[(size_t)bm * TK + tid];
        sIdx[tid] = k;
        sW[tid]   = edg[(size_t)bm * TK + tid];
        sSt[tid]  = srcState ? srcState[(size_t)b * srcStBS + k] : 0.0f;
    }
    __syncthreads();

    float a0 = baseVal[(size_t)b * baseBS + (size_t)m * DD + tid];
    float a1 = baseVal[(size_t)b * baseBS + (size_t)m * DD + 256 + tid];
#pragma unroll 4
    for (int k = 0; k < TK; ++k) {
        const float* row = srcVal + (size_t)b * srcBS + (size_t)sIdx[k] * DD;
        __builtin_prefetch(srcVal + (size_t)b * srcBS + (size_t)sIdx[(k + 1) & (TK - 1)] * DD + tid, 0, 1);
        const float w = sW[k];
        a0 = fmaf(w, row[tid], a0);
        a1 = fmaf(w, row[256 + tid], a1);
    }

    float sum = block_reduce_sum(a0 + a1, red, tid);
    if (tid == 0) sMu = sum * (1.0f / DD);
    __syncthreads();
    const float mu = sMu;
    const float d0 = a0 - mu, d1 = a1 - mu;
    float s2 = block_reduce_sum(d0 * d0 + d1 * d1, red, tid);
    if (tid == 0) sRs = rsqrtf(s2 * (1.0f / DD) + LNEPS);
    __syncthreads();
    const float rs = sRs;

    const size_t o = (size_t)bm * DD;
    outVal[o + tid]       = d0 * rs * g[tid]       + bvec[tid];
    outVal[o + 256 + tid] = d1 * rs * g[256 + tid] + bvec[256 + tid];

    if (outState && tid == 0) {
        float st = baseState[(size_t)b * baseStBS + m];
#pragma unroll
        for (int k = 0; k < TK; ++k) st = fmaf(sW[k], sSt[k], st);
        outState[bm] = st;
    }
}

// ---------------------------------------------------------------------------
extern "C" void kernel_launch(void* const* d_in, const int* in_sizes, int n_in,
                              void* d_out, int out_size, void* d_ws, size_t ws_size,
                              hipStream_t stream) {
    const float* token_val   = (const float*)d_in[0];
    const float* token_state = (const float*)d_in[1];
    const float* init_state  = (const float*)d_in[2];
    const float* init_val    = (const float*)d_in[3];
    const float* rUs_w = (const float*)d_in[4];
    const float* rUs_b = (const float*)d_in[5];
    const float* rUt_w = (const float*)d_in[6];
    const float* rUt_b = (const float*)d_in[7];
    const float* r_w   = (const float*)d_in[8];
    const float* pUs_w = (const float*)d_in[9];
    const float* pUs_b = (const float*)d_in[10];
    const float* pUt_w = (const float*)d_in[11];
    const float* pUt_b = (const float*)d_in[12];
    const float* p_w   = (const float*)d_in[13];
    const float* ln_g  = (const float*)d_in[14];
    const float* ln_b  = (const float*)d_in[15];
    (void)in_sizes; (void)n_in; (void)out_size; (void)ws_size;

    float* ws = (float*)d_ws;
    float* mem_val0 = ws; ws += (size_t)MM * DD;       // LN(init_val), [M,D]
    float* state0   = ws; ws += MM;                    // [M]
    float* ps       = ws; ws += (size_t)BB * SS * RR;  // token proj, scaled
    float* pt       = ws; ws += (size_t)MM * RR;       // slot proj (batch-shared)
    int*   idx1     = (int*)ws; ws += (size_t)BB * MM * TK;
    float* edg1     = ws; ws += (size_t)BB * MM * TK;
    float* stRaw    = ws; ws += (size_t)BB * MM;
    float* mem_val1 = ws; ws += (size_t)BB * MM * DD;  // post-write memory
    float* state1   = ws; ws += (size_t)BB * MM;
    float* pt2      = ws; ws += (size_t)BB * MM * RR;
    float* ps2      = ws; ws += (size_t)BB * MM * RR;
    int*   idx2     = (int*)ws; ws += (size_t)BB * MM * TK;
    float* edg2     = ws; ws += (size_t)BB * MM * TK;

    // 1) base memory
    ln_rows<<<MM, 256, 0, stream>>>(init_val, ln_g, ln_b, mem_val0);
    signed_softmax_mass<<<1, 256, 0, stream>>>(init_state, state0);

    // 2) low-rank projections for write routing (scale r_w*0.1 folded into ps)
    proj_gemm<<<MM / 16, 32, 0, stream>>>(mem_val0, rUt_w, rUt_b, nullptr, nullptr, 1.0f, pt);
    proj_gemm<<<(BB * SS) / 16, 32, 0, stream>>>(token_val, rUs_w, rUs_b, r_w, nullptr, LRS, ps);

    // 3) fused score + top-16 + edge softmax (dst=slots, src=tokens)
    score_topk<<<dim3(MM / 16, BB), 32, 0, stream>>>(pt, 0L, ps, (long)SS * RR, SS, idx1, edg1);

    // 4) write: gather token rows, accumulate, LN; state raw update
    apply_ln<<<BB * MM, 256, 0, stream>>>(mem_val0, 0L, token_val, (long)SS * DD,
                                          state0, 0L, token_state, (long)SS,
                                          idx1, edg1, ln_g, ln_b, mem_val1, stRaw);
    signed_softmax_mass<<<BB, 256, 0, stream>>>(stRaw, state1);

    // 5) propagation projections (state column-weighting folded into ps2)
    proj_gemm<<<(BB * MM) / 16, 32, 0, stream>>>(mem_val1, pUt_w, pUt_b, nullptr, nullptr, 1.0f, pt2);
    proj_gemm<<<(BB * MM) / 16, 32, 0, stream>>>(mem_val1, pUs_w, pUs_b, p_w, state1, LRS, ps2);

    // 6) fused score + top-16 + edge softmax (slot<->slot)
    score_topk<<<dim3(MM / 16, BB), 32, 0, stream>>>(pt2, (long)MM * RR, ps2, (long)MM * RR, MM,
                                                     idx2, edg2);

    // 7) residual propagate + final LN -> output [B,M,D]
    apply_ln<<<BB * MM, 256, 0, stream>>>(mem_val1, (long)MM * DD, mem_val1, (long)MM * DD,
                                          nullptr, 0L, nullptr, 0L,
                                          idx2, edg2, ln_g, ln_b, (float*)d_out, nullptr);
}
